// AutoEncoder_1700807049822
// MI455X (gfx1250) — matrix operand order repaired and checked by reading, not compiled, not run on hardware
//
#include <hip/hip_runtime.h>
#include <math.h>

#define B_   256
#define S_   128
#define E_   768
#define E4_  (E_ / 4)     // 192 float4 per row
#define A_   30
#define NNEG 10

typedef __attribute__((ext_vector_type(16))) _Float16 v16h;
typedef __attribute__((ext_vector_type(8)))  float    v8f;

// ---------- helpers (wave32) ----------
__device__ __forceinline__ float wave_reduce_sum(float v) {
    #pragma unroll
    for (int o = 16; o > 0; o >>= 1) v += __shfl_xor(v, o, 32);
    return v;
}

// block-wide sum for blockDim.x == 256 (8 waves); returns the sum to all threads
__device__ __forceinline__ float block_reduce_sum(float v, float* red) {
    v = wave_reduce_sum(v);
    int w = threadIdx.x >> 5, l = threadIdx.x & 31;
    __syncthreads();                 // protect previous red[] consumers
    if (l == 0) red[w] = v;
    __syncthreads();
    if (threadIdx.x == 0) {
        float s = 0.f;
        #pragma unroll
        for (int i = 0; i < 8; ++i) s += red[i];
        red[0] = s;
    }
    __syncthreads();
    return red[0];
}

// ---------- K1: q = mean_s x  (pass 1 over x; warms L2). b128 streaming. ----------
__global__ void k_mean(const float* __restrict__ x, float* __restrict__ q) {
    int b = blockIdx.x, t = threadIdx.x;
    if (t >= E4_) return;
    const float4* xb = (const float4*)(x + (size_t)b * S_ * E_);
    float4 acc = make_float4(0.f, 0.f, 0.f, 0.f);
    for (int s = 0; s < S_; ++s) {
        float4 v = xb[s * E4_ + t];
        acc.x += v.x; acc.y += v.y; acc.z += v.z; acc.w += v.w;
    }
    const float inv = 1.0f / S_;
    acc.x *= inv; acc.y *= inv; acc.z *= inv; acc.w *= inv;
    ((float4*)q)[b * E4_ + t] = acc;
}

// ---------- K2: qW = q @ W_att + b_att via V_WMMA_F32_16X16X32_F16 ----------
// One wave per 16x16 tile. M=256 -> 16 tiles, N=768 -> 48 tiles, 768 waves total.
__global__ void k_qw_wmma(const float* __restrict__ q, const float* __restrict__ W,
                          const float* __restrict__ bias, float* __restrict__ qW) {
    int wave = threadIdx.x >> 5;
    int lane = threadIdx.x & 31;
    int tile = blockIdx.x * 8 + wave;     // 96 blocks * 8 waves = 768 tiles
    int mt = tile / 48, nt = tile % 48;

    int mrow = mt * 16 + (lane & 15);     // A: lane -> M
    int ncol = nt * 16 + (lane & 15);     // B/D: lane -> N
    v8f c = {};

    for (int kk = 0; kk < E_; kk += 32) {
        v16h a, bm;
        // A (16x32 f16): lanes 0-15 -> K = kk..kk+15, lanes 16-31 -> K = kk+16..kk+31
        const float* qrow = q + mrow * E_ + kk + ((lane < 16) ? 0 : 16);
        #pragma unroll
        for (int j = 0; j < 8; ++j) {
            a[j]     = (_Float16)qrow[j];
            a[j + 8] = (_Float16)qrow[8 + j];
        }
        // B (32x16 f16): lanes 0-15 -> K = kk..kk+15, lanes 16-31 -> K = kk+16..kk+31
        int kb = kk + ((lane < 16) ? 0 : 16);
        #pragma unroll
        for (int j = 0; j < 16; ++j) bm[j] = (_Float16)W[(kb + j) * E_ + ncol];

        c = __builtin_amdgcn_wmma_f32_16x16x32_f16(
                /*neg_a=*/false, a, /*neg_b=*/false, bm,
                /*c_mod=*/(short)0, c, /*reuse_a=*/false, /*reuse_b=*/false);
    }
    // D layout: VGPR r -> row (r | 8*laneHi), col = lane&15
    int mbase = mt * 16 + ((lane < 16) ? 0 : 8);
    float bv = bias[ncol];
    #pragma unroll
    for (int r = 0; r < 8; ++r) qW[(mbase + r) * E_ + ncol] = c[r] + bv;
}

// ---------- K3: fused attention + reduction + reconstruction, one block per b ----------
__global__ void k_attn(const float* __restrict__ x, const float* __restrict__ qW,
                       const float* __restrict__ W_red, const float* __restrict__ b_red,
                       const float* __restrict__ aspect_W,
                       float* __restrict__ zs_out, float* __restrict__ rec_out) {
    __shared__ float4 buf4[E4_];   // qW row, later reused as z_s row
    __shared__ float sc[S_];
    __shared__ float comp[A_];
    __shared__ float stat[2];
    float* buf = (float*)buf4;

    int b = blockIdx.x, t = threadIdx.x;
    int w = t >> 5, l = t & 31;
    const float4* xb4 = (const float4*)(x + (size_t)b * S_ * E_);

    if (t < E4_) buf4[t] = ((const float4*)qW)[b * E4_ + t];
    __syncthreads();

    // scores[s] = x[b,s,:] . qW[b,:]  (one wave per s, b128 lanes stride the row)
    for (int s = w; s < S_; s += 8) {
        float acc = 0.f;
        #pragma unroll
        for (int k = 0; k < E4_ / 32; ++k) {       // 6 float4 per lane
            int e4 = l + 32 * k;
            float4 xv = xb4[s * E4_ + e4];
            float4 qv = buf4[e4];
            acc += xv.x * qv.x + xv.y * qv.y + xv.z * qv.z + xv.w * qv.w;
        }
        acc = wave_reduce_sum(acc);
        if (l == 0) sc[s] = acc;
    }
    __syncthreads();

    // softmax over S (serial scalar parts are 128 iters — negligible)
    if (t == 0) { float mx = sc[0]; for (int s = 1; s < S_; ++s) mx = fmaxf(mx, sc[s]); stat[0] = mx; }
    __syncthreads();
    if (t < S_) sc[t] = __expf(sc[t] - stat[0]);
    __syncthreads();
    if (t == 0) { float sm = 0.f; for (int s = 0; s < S_; ++s) sm += sc[s]; stat[1] = 1.0f / sm; }
    __syncthreads();
    if (t < S_) sc[t] *= stat[1];
    __syncthreads();

    // z_s = attn @ x[b]  (float4 accumulators, b128 loads/stores)
    float4 zacc = make_float4(0.f, 0.f, 0.f, 0.f);
    if (t < E4_) {
        for (int s = 0; s < S_; ++s) {
            float a = sc[s];
            float4 xv = xb4[s * E4_ + t];
            zacc.x += a * xv.x; zacc.y += a * xv.y; zacc.z += a * xv.z; zacc.w += a * xv.w;
        }
        ((float4*)zs_out)[(size_t)b * E4_ + t] = zacc;
    }
    __syncthreads();               // buf4 free to reuse (readers done above)
    if (t < E4_) buf4[t] = zacc;
    __syncthreads();

    // composition logits = z . W_red[:,a] + b_red
    if (t < A_) {
        float acc = b_red[t];
        for (int k = 0; k < E_; ++k) acc += buf[k] * W_red[k * A_ + t];
        comp[t] = acc;
    }
    __syncthreads();
    if (t == 0) {
        float mx = comp[0];
        for (int a = 1; a < A_; ++a) mx = fmaxf(mx, comp[a]);
        float sm = 0.f;
        for (int a = 0; a < A_; ++a) { comp[a] = __expf(comp[a] - mx); sm += comp[a]; }
        float inv = 1.0f / sm;
        for (int a = 0; a < A_; ++a) comp[a] *= inv;
    }
    __syncthreads();

    // reconstructed = comp @ aspect_W  (float4)
    if (t < E4_) {
        float4 racc = make_float4(0.f, 0.f, 0.f, 0.f);
        const float4* aw4 = (const float4*)aspect_W;
        #pragma unroll
        for (int a = 0; a < A_; ++a) {
            float cv = comp[a];
            float4 av = aw4[a * E4_ + t];
            racc.x += cv * av.x; racc.y += cv * av.y; racc.z += cv * av.z; racc.w += cv * av.w;
        }
        ((float4*)rec_out)[(size_t)b * E4_ + t] = racc;
    }
}

// ---------- K4: loss slot zero (graph-replay safe) ----------
__global__ void k_zero(float* loss) { if (threadIdx.x == 0 && blockIdx.x == 0) *loss = 0.f; }

// ---------- K5: max-margin loss, one block per b ----------
__global__ void k_margin(const float* __restrict__ zs, const float* __restrict__ rec,
                         const int* __restrict__ negidx, float* __restrict__ loss) {
    __shared__ float4 recl4[E4_];
    __shared__ float red[8];
    int b = blockIdx.x, t = threadIdx.x;
    const float4* rb4 = (const float4*)(rec + (size_t)b * E_);

    float n2 = 0.f;
    if (t < E4_) {
        float4 v = rb4[t];
        recl4[t] = v;
        n2 = v.x * v.x + v.y * v.y + v.z * v.z + v.w * v.w;
    }
    n2 = block_reduce_sum(n2, red);
    float denom = fmaxf(sqrtf(n2), 1e-12f);   // F.normalize eps

    const float4* zb4 = (const float4*)(zs + (size_t)b * E_);
    float p = 0.f;
    if (t < E4_) {
        float4 z = zb4[t], r = recl4[t];
        p = z.x * r.x + z.y * r.y + z.z * r.z + z.w * r.w;
    }
    p = block_reduce_sum(p, red) / denom;

    float acc = 0.f;
    for (int n = 0; n < NNEG; ++n) {
        int j = negidx[b * NNEG + n];
        const float4* zj4 = (const float4*)(zs + (size_t)j * E_);
        float d = 0.f;
        if (t < E4_) {
            float4 z = zj4[t], r = recl4[t];
            d = z.x * r.x + z.y * r.y + z.z * r.z + z.w * r.w;
        }
        d = block_reduce_sum(d, red) / denom;
        if (t == 0) acc += fmaxf(1.0f - p + d, 0.f);
    }
    if (t == 0) atomicAdd(loss, acc * (1.0f / (float)(B_ * NNEG)));
}

// ---------- K6: orthogonality regularizer (single block) ----------
__global__ void k_reg(const float* __restrict__ aw, float* __restrict__ loss) {
    __shared__ float norms[A_];
    __shared__ float red[8];
    int t = threadIdx.x;
    if (t < A_) {
        float s = 0.f;
        const float* r = aw + t * E_;
        for (int e = 0; e < E_; ++e) s += r[e] * r[e];
        norms[t] = fmaxf(sqrtf(s), 1e-12f);
    }
    __syncthreads();
    float acc = 0.f;
    const int NP = (A_ * (A_ + 1)) / 2;      // 465 upper-triangular pairs
    for (int p = t; p < NP; p += 256) {
        int i = 0, rem = p;
        while (rem >= A_ - i) { rem -= A_ - i; ++i; }
        int j = i + rem;
        const float4* ri = (const float4*)(aw + i * E_);
        const float4* rj = (const float4*)(aw + j * E_);
        float d = 0.f;
        for (int e = 0; e < E4_; ++e) {
            float4 a = ri[e], c = rj[e];
            d += a.x * c.x + a.y * c.y + a.z * c.z + a.w * c.w;
        }
        float g = d / (norms[i] * norms[j]) - ((i == j) ? 1.0f : 0.0f);
        acc += g * g * ((i == j) ? 1.0f : 2.0f);
    }
    acc = block_reduce_sum(acc, red);
    if (t == 0) atomicAdd(loss, sqrtf(acc));
}

extern "C" void kernel_launch(void* const* d_in, const int* in_sizes, int n_in,
                              void* d_out, int out_size, void* d_ws, size_t ws_size,
                              hipStream_t stream) {
    const float* x        = (const float*)d_in[0];
    const float* W_att    = (const float*)d_in[1];
    const float* b_att    = (const float*)d_in[2];
    const float* W_red    = (const float*)d_in[3];
    const float* b_red    = (const float*)d_in[4];
    const float* aspect_W = (const float*)d_in[5];
    const int*   neg_idx  = (const int*)d_in[6];

    float* out  = (float*)d_out;
    float* zs   = out;                               // [B,E]
    float* rec  = out + (size_t)B_ * E_;             // [B,E]
    float* loss = out + (size_t)2 * B_ * E_;         // [1]

    float* q  = (float*)d_ws;                        // [B,E]
    float* qW = q + (size_t)B_ * E_;                 // [B,E]

    k_mean   <<<B_, 256, 0, stream>>>(x, q);
    k_qw_wmma<<<96, 256, 0, stream>>>(q, W_att, b_att, qW);
    k_attn   <<<B_, 256, 0, stream>>>(x, qW, W_red, b_red, aspect_W, zs, rec);
    k_zero   <<<1, 1, 0, stream>>>(loss);
    k_margin <<<B_, 256, 0, stream>>>(zs, rec, neg_idx, loss);
    k_reg    <<<1, 256, 0, stream>>>(aspect_W, loss);
}
